// SCAM_73151882985874
// MI455X (gfx1250) — matrix operand order repaired and checked by reading, not compiled, not run on hardware
//
#include <hip/hip_runtime.h>

typedef __attribute__((ext_vector_type(16))) _Float16 v16h;
typedef __attribute__((ext_vector_type(8)))  float    v8f;

union HFrag { v16h v; _Float16 h[16]; unsigned int u[8]; };

// A[m][k] = base[m*ldm + k]  (16x32 f16 A-fragment, ISA 7.12.2 layout)
__device__ __forceinline__ v16h load_fragA(const _Float16* base, int ldm, int lane) {
    HFrag f;
    const int m  = lane & 15;
    const int hi = lane >> 4;
#pragma unroll
    for (int r = 0; r < 8; ++r) {
        const int k = ((r >> 2) << 4) + (hi << 3) + ((r & 3) << 1);
        f.u[r] = *reinterpret_cast<const unsigned int*>(base + m * ldm + k);
    }
    return f.v;
}

// B[k][n] = base[n*ldm + k]  (32x16 f16 B-fragment; lane = N, K split across halves)
__device__ __forceinline__ v16h load_fragB(const _Float16* base, int ldm, int lane) {
    HFrag f;
    const int n  = lane & 15;
    const int hi = lane >> 4;
#pragma unroll
    for (int r = 0; r < 8; ++r) {
        const int k = ((r >> 2) << 4) + (hi << 3) + ((r & 3) << 1);
        f.u[r] = *reinterpret_cast<const unsigned int*>(base + n * ldm + k);
    }
    return f.v;
}

// LDS strides (halves / words). Odd-word strides -> conflict-free access.
constexpr int QS = 66;    // Q/S tiles: 256 x 64 f16, stride 66
constexpr int RS = 258;   // R/L tiles (value, c-major): 64 x 256 f16, stride 258
constexpr int SS = 258;   // score: 256 x 256 f16, stride 258
constexpr int WS = 66;    // f16 weight tiles in LDS: 64 x {66|64}, stride 66
constexpr int OS = 65;    // f32 output staging: 256 x 64, stride 65 words
constexpr size_t LDS_BYTES =
    (size_t)(2 * 256 * QS + 2 * 64 * RS + 256 * SS) * 2 + 4 * 256 * 4;  // 269,824

__global__ __launch_bounds__(256)
void scam_fused(const float* __restrict__ x,      const float* __restrict__ re,
                const float* __restrict__ tr,
                const float* __restrict__ Wl1,    const float* __restrict__ bl1,
                const float* __restrict__ Wr1,    const float* __restrict__ br1,
                const float* __restrict__ Wl2,    const float* __restrict__ bl2,
                const float* __restrict__ Wr2,    const float* __restrict__ br2,
                const float* __restrict__ beta1,  const float* __restrict__ gamma1,
                const float* __restrict__ beta3,  const float* __restrict__ gamma3,
                float* __restrict__ out) {
    extern __shared__ char smem[];
    _Float16* Qs  = (_Float16*)smem;          // [i][c]  256 x QS
    _Float16* Ssh = Qs  + 256 * QS;           // [j][c]  256 x QS
    _Float16* Rsh = Ssh + 256 * QS;           // [c][j]   64 x RS  (bufR^T)
    _Float16* Lsh = Rsh + 64 * RS;            // [c][j]   64 x RS  (bufL^T)
    _Float16* sc  = Lsh + 64 * RS;            // [i][j]  256 x SS  score / rowExp
    float* rowMax = (float*)(sc + 256 * SS);
    float* rowInv = rowMax + 256;
    float* colMax = rowInv + 256;
    float* colInv = colMax + 256;
    // Overlays (lifetimes disjoint with their hosts):
    float*    outStage = (float*)smem;        // 256 x OS f32, over Qs+Ssh (dead after phase A)
    _Float16* xl16 = sc;                      // f16 x_l rows, over sc (dead before phase A)
    _Float16* xr16 = sc + 256 * QS;
    _Float16* w1l  = sc + 2 * 256 * QS;       // Wl1 f16 [64][66]
    _Float16* w1r  = w1l + 64 * WS;           // Wr1 f16 [64][66]
    _Float16* w2l  = w1r + 64 * WS;           // Wl2 f16 [64][66] (cols 64.. pad)
    _Float16* w2r  = w2l + 64 * WS;           // Wr2 f16 [64][66]

    const int bh   = blockIdx.x;
    const int b    = bh >> 8;       // 0..1 left batch; right batch = b+2
    const int h    = bh & 255;
    const int tid  = threadIdx.x;
    const int lane = tid & 31;
    const int wave = tid >> 5;
    const size_t PL = 65536;        // channel plane stride (h*w)
    const size_t basel = (size_t)b       * 64 * PL + (size_t)h * 256 + tid;
    const size_t baser = (size_t)(b + 2) * 64 * PL + (size_t)h * 256 + tid;

    // ---- Phase 0a: stage x (f32->f16) and weights (f32->f16) into LDS ----
    {
        const int i = tid;
#pragma unroll 4
        for (int ci = 0; ci < 64; ++ci) {
            xl16[i * QS + ci] = (_Float16)x[basel + (size_t)ci * PL];
            xr16[i * QS + ci] = (_Float16)x[baser + (size_t)ci * PL];
        }
#pragma unroll 1
        for (int idx = tid; idx < 64 * 66; idx += 256) {
            w1l[idx] = (_Float16)Wl1[idx];
            w1r[idx] = (_Float16)Wr1[idx];
            const int co = idx / 66, ci = idx - co * 66;
            w2l[idx] = (_Float16)((ci < 64) ? Wl2[co * 64 + ci] : 0.0f);
            w2r[idx] = (_Float16)((ci < 64) ? Wr2[co * 64 + ci] : 0.0f);
        }
    }
    __syncthreads();

    // ---- Phase 0b: WMMA 1x1 convs (K=64) -> Qs, Ssh, Lsh, Rsh -------------
    const float yy = (float)h * (2.0f / 255.0f) - 1.0f;
    const float rr = fabsf(yy - 0.5f);
    for (int t = wave; t < 64; t += 8) {          // 16 pixel-tiles x 4 cout-tiles
        const int ti = t >> 2, tc = t & 3;
        const int i0 = ti << 4, c0 = tc << 4;
        const int n = lane & 15, hi = lane >> 4, mh = hi << 3;
        const int co = c0 + n;
        const v16h al0 = load_fragA(xl16 + i0 * QS,      QS, lane);
        const v16h al1 = load_fragA(xl16 + i0 * QS + 32, QS, lane);
        const v16h ar0 = load_fragA(xr16 + i0 * QS,      QS, lane);
        const v16h ar1 = load_fragA(xr16 + i0 * QS + 32, QS, lane);
        {   // Q = CoordConv(x_l; Wl1): coord+bias folded into accumulator init
            const float bias = bl1[co] + (float)w1l[co * WS + 64] * yy
                                       + (float)w1l[co * WS + 65] * rr;
            v8f acc;
#pragma unroll
            for (int r = 0; r < 8; ++r) acc[r] = bias;
            acc = __builtin_amdgcn_wmma_f32_16x16x32_f16(false, al0, false,
                    load_fragB(w1l + c0 * WS,      WS, lane), (short)0, acc, false, false);
            acc = __builtin_amdgcn_wmma_f32_16x16x32_f16(false, al1, false,
                    load_fragB(w1l + c0 * WS + 32, WS, lane), (short)0, acc, false, false);
#pragma unroll
            for (int r = 0; r < 8; ++r) Qs[(i0 + r + mh) * QS + co] = (_Float16)acc[r];
        }
        {   // S = CoordConv(x_r; Wr1)
            const float bias = br1[co] + (float)w1r[co * WS + 64] * yy
                                       + (float)w1r[co * WS + 65] * rr;
            v8f acc;
#pragma unroll
            for (int r = 0; r < 8; ++r) acc[r] = bias;
            acc = __builtin_amdgcn_wmma_f32_16x16x32_f16(false, ar0, false,
                    load_fragB(w1r + c0 * WS,      WS, lane), (short)0, acc, false, false);
            acc = __builtin_amdgcn_wmma_f32_16x16x32_f16(false, ar1, false,
                    load_fragB(w1r + c0 * WS + 32, WS, lane), (short)0, acc, false, false);
#pragma unroll
            for (int r = 0; r < 8; ++r) Ssh[(i0 + r + mh) * QS + co] = (_Float16)acc[r];
        }
        {   // bufL^T = conv(x_l; Wl2), stored [c][i]
            v8f acc;
#pragma unroll
            for (int r = 0; r < 8; ++r) acc[r] = bl2[co];
            acc = __builtin_amdgcn_wmma_f32_16x16x32_f16(false, al0, false,
                    load_fragB(w2l + c0 * WS,      WS, lane), (short)0, acc, false, false);
            acc = __builtin_amdgcn_wmma_f32_16x16x32_f16(false, al1, false,
                    load_fragB(w2l + c0 * WS + 32, WS, lane), (short)0, acc, false, false);
#pragma unroll
            for (int r = 0; r < 8; ++r) Lsh[co * RS + i0 + r + mh] = (_Float16)acc[r];
        }
        {   // bufR^T = conv(x_r; Wr2), stored [c][j]
            v8f acc;
#pragma unroll
            for (int r = 0; r < 8; ++r) acc[r] = br2[co];
            acc = __builtin_amdgcn_wmma_f32_16x16x32_f16(false, ar0, false,
                    load_fragB(w2r + c0 * WS,      WS, lane), (short)0, acc, false, false);
            acc = __builtin_amdgcn_wmma_f32_16x16x32_f16(false, ar1, false,
                    load_fragB(w2r + c0 * WS + 32, WS, lane), (short)0, acc, false, false);
#pragma unroll
            for (int r = 0; r < 8; ++r) Rsh[co * RS + i0 + r + mh] = (_Float16)acc[r];
        }
    }
    __syncthreads();

    // ---- Phase 0c: += beta1*tr (Q,S) and += gamma1*re (L,R), coalesced ----
    {
        const int i = tid;
#pragma unroll 1
        for (int co = 0; co < 64; ++co) {
            const float b1 = beta1[co], g1 = gamma1[co];
            Qs [i * QS + co] = (_Float16)((float)Qs [i * QS + co] + b1 * tr[basel + (size_t)co * PL]);
            Ssh[i * QS + co] = (_Float16)((float)Ssh[i * QS + co] + b1 * tr[baser + (size_t)co * PL]);
            Lsh[co * RS + i] = (_Float16)((float)Lsh[co * RS + i] + g1 * re[basel + (size_t)co * PL]);
            Rsh[co * RS + i] = (_Float16)((float)Rsh[co * RS + i] + g1 * re[baser + (size_t)co * PL]);
        }
    }
    __syncthreads();

    // ---- Phase A: score = 0.125 * Q S^T with triangular mask (j>i) --------
    const _Float16 HMIN = (_Float16)(-65504.0f);
    for (int t = wave; t < 256; t += 8) {         // wave-uniform tile index
        const int ti = t >> 4, tj = t & 15;
        const int i0 = ti << 4, j0 = tj << 4;
        const int n = lane & 15, mh = (lane >> 4) << 3;
        if (tj > ti) {                            // fully masked tile
#pragma unroll
            for (int r = 0; r < 8; ++r) sc[(i0 + r + mh) * SS + (j0 + n)] = HMIN;
            continue;
        }
        v8f acc = {};
#pragma unroll
        for (int k0 = 0; k0 < 64; k0 += 32) {
            const v16h af = load_fragA(Qs  + i0 * QS + k0, QS, lane);
            const v16h bf = load_fragB(Ssh + j0 * QS + k0, QS, lane);
            acc = __builtin_amdgcn_wmma_f32_16x16x32_f16(false, af, false, bf,
                                                         (short)0, acc, false, false);
        }
#pragma unroll
        for (int r = 0; r < 8; ++r) {
            const int gi = i0 + r + mh, gj = j0 + n;
            const float v = (gj > gi) ? -65504.0f : acc[r] * 0.125f;   // 64^-0.5
            sc[gi * SS + gj] = (_Float16)v;
        }
    }
    __syncthreads();

    // ---- Phase B: row & column softmax statistics -------------------------
    {
        const int t = tid;
        float mx = -3.0e38f;
        for (int j = 0; j < 256; ++j) mx = fmaxf(mx, (float)sc[t * SS + j]);
        float s = 0.0f;
        for (int j = 0; j < 256; ++j) s += __expf((float)sc[t * SS + j] - mx);
        rowMax[t] = mx; rowInv[t] = 1.0f / s;

        float cm = -3.0e38f;
        for (int i2 = 0; i2 < 256; ++i2) cm = fmaxf(cm, (float)sc[i2 * SS + t]);
        float cs = 0.0f;
        for (int i2 = 0; i2 < 256; ++i2) cs += __expf((float)sc[i2 * SS + t] - cm);
        colMax[t] = cm; colInv[t] = 1.0f / cs;
    }
    __syncthreads();

    // ---- Phase D: buf_r = M_l2r @ bufL^T (transposed softmax, exp on-the-fly)
    for (int t = wave; t < 64; t += 8) {
        const int ta = t >> 2, tc = t & 3;
        const int a0 = ta << 4, c0 = tc << 4;
        const int m = lane & 15, hi = lane >> 4;
        const float cmx = colMax[a0 + m];
        v8f acc = {};
#pragma unroll 1
        for (int k0 = 0; k0 < 256; k0 += 32) {
            HFrag a;
#pragma unroll
            for (int r = 0; r < 8; ++r) {
                const int k = ((r >> 2) << 4) + (hi << 3) + ((r & 3) << 1);
                a.h[2 * r]     = (_Float16)__expf((float)sc[(k0 + k)     * SS + a0 + m] - cmx);
                a.h[2 * r + 1] = (_Float16)__expf((float)sc[(k0 + k + 1) * SS + a0 + m] - cmx);
            }
            const v16h bf = load_fragB(Lsh + c0 * RS + k0, RS, lane);
            acc = __builtin_amdgcn_wmma_f32_16x16x32_f16(false, a.v, false, bf,
                                                         (short)0, acc, false, false);
        }
        const int n = lane & 15, mh = hi << 3;
#pragma unroll
        for (int r = 0; r < 8; ++r) {
            const int ga = a0 + r + mh;
            outStage[ga * OS + c0 + n] = acc[r] * colInv[ga];
        }
    }
    __syncthreads();

    // ---- Drain D (coalesced out_R) + in-place rowExp conversion of sc -----
    {
        const int i = tid;
#pragma unroll 1
        for (int co = 0; co < 64; ++co) {
            const size_t o = (((size_t)(b + 2) * 64 + co) * 256 + h) * 256 + i;
            out[o] = x[o] + gamma3[co] * outStage[i * OS + co];
        }
        const float rm = rowMax[i];
#pragma unroll 1
        for (int j = 0; j < 256; ++j)
            sc[i * SS + j] = (_Float16)__expf((float)sc[i * SS + j] - rm);
    }
    __syncthreads();

    // ---- Phase C: buf_l = rowExp @ bufR^T (A-frags are plain ds loads) ----
    for (int t = wave; t < 64; t += 8) {
        const int ti = t >> 2, tc = t & 3;
        const int i0 = ti << 4, c0 = tc << 4;
        v8f acc = {};
#pragma unroll 1
        for (int k0 = 0; k0 < 256; k0 += 32) {
            const v16h af = load_fragA(sc + i0 * SS + k0, SS, lane);
            const v16h bf = load_fragB(Rsh + c0 * RS + k0, RS, lane);
            acc = __builtin_amdgcn_wmma_f32_16x16x32_f16(false, af, false, bf,
                                                         (short)0, acc, false, false);
        }
        const int n = lane & 15, mh = (lane >> 4) << 3;
#pragma unroll
        for (int r = 0; r < 8; ++r) {
            const int gi = i0 + r + mh;
            outStage[gi * OS + c0 + n] = acc[r] * rowInv[gi];
        }
    }
    __syncthreads();

    // ---- Drain C (coalesced out_L) ----------------------------------------
    {
        const int i = tid;
#pragma unroll 1
        for (int co = 0; co < 64; ++co) {
            const size_t o = (((size_t)b * 64 + co) * 256 + h) * 256 + i;
            out[o] = x[o] + beta3[co] * outStage[i * OS + co];
        }
    }
}

extern "C" void kernel_launch(void* const* d_in, const int* in_sizes, int n_in,
                              void* d_out, int out_size, void* d_ws, size_t ws_size,
                              hipStream_t stream) {
    (void)in_sizes; (void)n_in; (void)out_size; (void)d_ws; (void)ws_size;
    const float* x      = (const float*)d_in[0];
    const float* re     = (const float*)d_in[1];
    const float* tr     = (const float*)d_in[2];
    const float* Wl1    = (const float*)d_in[3];
    const float* bl1    = (const float*)d_in[4];
    const float* Wr1    = (const float*)d_in[5];
    const float* br1    = (const float*)d_in[6];
    const float* Wl2    = (const float*)d_in[7];
    const float* bl2    = (const float*)d_in[8];
    const float* Wr2    = (const float*)d_in[9];
    const float* br2    = (const float*)d_in[10];
    const float* beta1  = (const float*)d_in[11];
    const float* gamma1 = (const float*)d_in[12];
    const float* beta3  = (const float*)d_in[13];
    const float* gamma3 = (const float*)d_in[14];
    float* out = (float*)d_out;

    (void)hipFuncSetAttribute((const void*)scam_fused,
                              hipFuncAttributeMaxDynamicSharedMemorySize,
                              (int)LDS_BYTES);
    // one workgroup per (batch-pair, image row): 2 * 256 = 512 blocks, 8 waves each
    scam_fused<<<dim3(512), dim3(256), LDS_BYTES, stream>>>(
        x, re, tr, Wl1, bl1, Wr1, br1, Wl2, bl2, Wr2, br2,
        beta1, gamma1, beta3, gamma3, out);
}
